// Curvphormer_90623809946318
// MI455X (gfx1250) — matrix operand order repaired and checked by reading, not compiled, not run on hardware
//
#include <hip/hip_runtime.h>
#include <math.h>

#define N_NODES 50000
#define N_EDGES 625000
#define D 128
#define H 8
#define L 4
#define F_IN 64
#define G 64

#define LDS_STRIDE 40   // halfs; 80B row stride -> bank-conflict-free frag loads, 16B aligned

typedef __attribute__((ext_vector_type(16))) _Float16 v16h;
typedef __attribute__((ext_vector_type(8)))  float    v8f;

union FragU { uint4 u[2]; v16h v; };
union Pack4 { _Float16 h[4]; uint2 u; };

// ---------------- helpers ----------------
__device__ __forceinline__ unsigned int enc_f32(float f) {
  unsigned int b = __float_as_uint(f);
  return (b & 0x80000000u) ? ~b : (b | 0x80000000u);  // order-preserving float->uint
}
__device__ __forceinline__ float dec_f32(unsigned int u) {
  unsigned int b = (u & 0x80000000u) ? (u ^ 0x80000000u) : ~u;
  return __uint_as_float(b);
}

// ---------------- WMMA GEMM: C[M,N] = A[M,K]@B[K,N] (+bias)(+resid)(relu) ----------------
// fp32 in/out, fp16 WMMA compute, fp32 accumulate.
// Block = 256 thr = 8 waves; block tile 128x64; each wave owns 16 rows x 64 cols
// (4 accumulators), reusing one A fragment across 4 B fragments -> 4 WMMA / stage.
__global__ void __launch_bounds__(256)
wmma_gemm_kernel(const float* __restrict__ A, const float* __restrict__ B,
                 const float* __restrict__ bias, const float* __restrict__ resid,
                 float* __restrict__ C, int M, int K, int N, int flags)
{
  __shared__ __align__(16) _Float16 As[128 * LDS_STRIDE];  // [row][k] fp16, padded stride
  __shared__ __align__(16) _Float16 Bs[64 * LDS_STRIDE];   // [n][k]  fp16 (transposed stage)

  const int tid  = threadIdx.x;
  const int lane = tid & 31;
  const int wid  = tid >> 5;            // 0..7 -> 16-row slab of the 128-row block tile
  const int m0 = blockIdx.y * 128;
  const int n0 = blockIdx.x * 64;

  v8f acc0 = {}, acc1 = {}, acc2 = {}, acc3 = {};

  const int acol = (tid & 7) * 4;       // 0,4,..,28 (k within tile)
  const int bn   = (tid & 15) * 4;      // 0..60
  const int bk   = tid >> 4;            // 0..15
  int aRowG[4];
#pragma unroll
  for (int i = 0; i < 4; ++i) {
    int r = (tid >> 3) + i * 32;        // 0..127
    aRowG[i] = (m0 + r < M) ? (m0 + r) : (M - 1);   // clamp tail rows
  }

  for (int k0 = 0; k0 < K; k0 += 32) {
    // ---- stage A (128x32 fp32 -> fp16): 4 float4 loads, 4 packed ds_store_b64 ----
#pragma unroll
    for (int i = 0; i < 4; ++i) {
      int row = (tid >> 3) + i * 32;
      float4 av = *(const float4*)(A + (size_t)aRowG[i] * K + k0 + acol);
      Pack4 pk;
      pk.h[0] = (_Float16)av.x; pk.h[1] = (_Float16)av.y;
      pk.h[2] = (_Float16)av.z; pk.h[3] = (_Float16)av.w;
      *(uint2*)(As + row * LDS_STRIDE + acol) = pk.u;
    }
    if (k0 + 32 < K)
      __builtin_prefetch(A + (size_t)aRowG[0] * K + k0 + 32 + acol, 0, 1); // global_prefetch_b8
    // ---- stage B (32x64 fp32 -> fp16, transposed to [n][k]) ----
#pragma unroll
    for (int p = 0; p < 2; ++p) {
      int kk = bk + p * 16;
      float4 bv = *(const float4*)(B + (size_t)(k0 + kk) * N + n0 + bn);
      Bs[(bn + 0) * LDS_STRIDE + kk] = (_Float16)bv.x;
      Bs[(bn + 1) * LDS_STRIDE + kk] = (_Float16)bv.y;
      Bs[(bn + 2) * LDS_STRIDE + kk] = (_Float16)bv.z;
      Bs[(bn + 3) * LDS_STRIDE + kk] = (_Float16)bv.w;
    }
    __syncthreads();

    // Fragment fetch in contiguous k order k = 16*half + s, not the ISA table order (16-bit A 16x32):
    //  lanes 0-15: V0-7 = K0..15 ; lanes 16-31: V0-7 = K16..31 (same k order as B)
    // B (32x16): lanes 0-15: V0-7 = K0..15 ; lanes 16-31: V0-7 = K16..31
    const int half = lane >> 4;
    const uint4* ar = (const uint4*)(As + (wid * 16 + (lane & 15)) * LDS_STRIDE);
    FragU fa;
    fa.u[0] = ar[half * 2];
    fa.u[1] = ar[half * 2 + 1];

    FragU fb0, fb1, fb2, fb3;
    {
      const uint4* br0 = (const uint4*)(Bs + (0 * 16 + (lane & 15)) * LDS_STRIDE);
      const uint4* br1 = (const uint4*)(Bs + (1 * 16 + (lane & 15)) * LDS_STRIDE);
      const uint4* br2 = (const uint4*)(Bs + (2 * 16 + (lane & 15)) * LDS_STRIDE);
      const uint4* br3 = (const uint4*)(Bs + (3 * 16 + (lane & 15)) * LDS_STRIDE);
      fb0.u[0] = br0[half * 2]; fb0.u[1] = br0[half * 2 + 1];
      fb1.u[0] = br1[half * 2]; fb1.u[1] = br1[half * 2 + 1];
      fb2.u[0] = br2[half * 2]; fb2.u[1] = br2[half * 2 + 1];
      fb3.u[0] = br3[half * 2]; fb3.u[1] = br3[half * 2 + 1];
    }

    acc0 = __builtin_amdgcn_wmma_f32_16x16x32_f16(false, fa.v, false, fb0.v, (short)0, acc0, false, false);
    acc1 = __builtin_amdgcn_wmma_f32_16x16x32_f16(false, fa.v, false, fb1.v, (short)0, acc1, false, false);
    acc2 = __builtin_amdgcn_wmma_f32_16x16x32_f16(false, fa.v, false, fb2.v, (short)0, acc2, false, false);
    acc3 = __builtin_amdgcn_wmma_f32_16x16x32_f16(false, fa.v, false, fb3.v, (short)0, acc3, false, false);
    __syncthreads();
  }

  // Epilogue: D layout — VGPR r: lanes0-15 -> M=r, lanes16-31 -> M=r+8, N=lane%16
  const int rbase = m0 + wid * 16 + ((lane >> 4) * 8);
  v8f accs[4] = {acc0, acc1, acc2, acc3};
#pragma unroll
  for (int j = 0; j < 4; ++j) {
    const int col = n0 + j * 16 + (lane & 15);
    const float bcol = bias ? bias[col] : 0.0f;
#pragma unroll
    for (int r = 0; r < 8; ++r) {
      int row = rbase + r;
      if (row < M) {
        float v = accs[j][r] + bcol;
        if (flags & 1) v = fmaxf(v, 0.0f);
        size_t idx = (size_t)row * N + col;
        if (flags & 2) v += resid[idx];
        C[idx] = v;
      }
    }
  }
}

// ---------------- LayerNorm over D=128, one wave per node ----------------
__global__ void __launch_bounds__(256)
layernorm_kernel(const float* __restrict__ x, const float* __restrict__ g,
                 const float* __restrict__ b, float* __restrict__ y, int M)
{
  int wid = threadIdx.x >> 5, lane = threadIdx.x & 31;
  int n = blockIdx.x * 8 + wid;
  if (n >= M) return;
  float4 v = *(const float4*)(x + (size_t)n * D + lane * 4);
  float s1 = v.x + v.y + v.z + v.w;
  float s2 = v.x * v.x + v.y * v.y + v.z * v.z + v.w * v.w;
  for (int m = 16; m >= 1; m >>= 1) {
    s1 += __shfl_xor(s1, m, 32);
    s2 += __shfl_xor(s2, m, 32);
  }
  float mu  = s1 * (1.0f / 128.0f);
  float var = s2 * (1.0f / 128.0f) - mu * mu;
  float inv = rsqrtf(var + 1e-5f);
  int d0 = lane * 4;
  float4 o;
  o.x = (v.x - mu) * inv * g[d0 + 0] + b[d0 + 0];
  o.y = (v.y - mu) * inv * g[d0 + 1] + b[d0 + 1];
  o.z = (v.z - mu) * inv * g[d0 + 2] + b[d0 + 2];
  o.w = (v.w - mu) * inv * g[d0 + 3] + b[d0 + 3];
  *(float4*)(y + (size_t)n * D + d0) = o;
}

// ---------------- Edge pass 1: sim, dist, segment-max(sim over tgt) ----------------
__global__ void __launch_bounds__(256)
edge_simdist_kernel(const float* __restrict__ h, const int* __restrict__ src,
                    const int* __restrict__ tgt, float* __restrict__ sim,
                    float* __restrict__ dist, unsigned int* __restrict__ smax, int E)
{
  int wid = threadIdx.x >> 5, lane = threadIdx.x & 31;
  int e = blockIdx.x * 8 + wid;
  if (e >= E) return;
  int s = src[e], t = tgt[e];
  float4 a = *(const float4*)(h + (size_t)s * D + lane * 4);
  float4 b = *(const float4*)(h + (size_t)t * D + lane * 4);
  float dot = a.x * b.x + a.y * b.y + a.z * b.z + a.w * b.w;
  float dx = a.x - b.x, dy = a.y - b.y, dz = a.z - b.z, dw = a.w - b.w;
  float dd = dx * dx + dy * dy + dz * dz + dw * dw;
  for (int m = 16; m >= 1; m >>= 1) {
    dot += __shfl_xor(dot, m, 32);
    dd  += __shfl_xor(dd, m, 32);
  }
  if (lane == 0) {
    float sv = dot;               // BETA = 1.0
    sim[e]  = sv;
    dist[e] = sqrtf(dd);
    atomicMax(&smax[t], enc_f32(sv));
  }
}

// exp(sim - max) and segment-sum of denom (thread per edge); sim overwritten with exp
__global__ void __launch_bounds__(256)
edge_expdenom_kernel(float* __restrict__ sim, const unsigned int* __restrict__ smax,
                     const int* __restrict__ tgt, float* __restrict__ denom, int E)
{
  int i = blockIdx.x * 256 + threadIdx.x;
  if (i >= E) return;
  int t = tgt[i];
  float ex = expf(sim[i] - dec_f32(smax[t]));
  sim[i] = ex;
  atomicAdd(&denom[t], ex);
}

// agg[tgt] += alpha*dist (thread per edge)
__global__ void __launch_bounds__(256)
edge_alphaagg_kernel(const float* __restrict__ expw, const float* __restrict__ dist,
                     const float* __restrict__ denom, const int* __restrict__ tgt,
                     float* __restrict__ aggd, int E)
{
  int i = blockIdx.x * 256 + threadIdx.x;
  if (i >= E) return;
  int t = tgt[i];
  float a = expw[i] / denom[t];
  atomicAdd(&aggd[t], a * dist[i]);
}

// Per-layer precompute: Mb = cW2 @ bW[l] (128x8), cconst = cb2 @ bW[l] + bb[l]
__global__ void prep_curv_proj_kernel(const float* __restrict__ cW2, const float* __restrict__ cb2,
                                      const float* __restrict__ bWl, const float* __restrict__ bbl,
                                      float* __restrict__ Mb, float* __restrict__ cconst)
{
  int tid = blockIdx.x * 256 + threadIdx.x;
  if (tid < D * H) {
    int j = tid >> 3, hh = tid & 7;
    float a = 0.0f;
    for (int d = 0; d < D; ++d) a += cW2[j * D + d] * bWl[d * H + hh];
    Mb[tid] = a;
  }
  if (tid < H) {
    float a = bbl[tid];
    for (int d = 0; d < D; ++d) a += cb2[d] * bWl[d * H + tid];
    cconst[tid] = a;
  }
}

// scores[e][h] = (q[src].k[tgt])_h/4 + relu(curv*cW1+cb1)@Mb + cconst; segment-max over tgt
__global__ void __launch_bounds__(256)
edge_scores_kernel(const float* __restrict__ q, const float* __restrict__ k,
                   const int* __restrict__ src, const int* __restrict__ tgt,
                   const float* __restrict__ aggd, const float* __restrict__ dist,
                   const float* __restrict__ cW1, const float* __restrict__ cb1,
                   const float* __restrict__ Mb, const float* __restrict__ cconst,
                   float* __restrict__ scores, unsigned int* __restrict__ smax, int E)
{
  __shared__ float sMb[D * H];
  __shared__ float sW1[D], sB1[D], sC[H];
  for (int i = threadIdx.x; i < D * H; i += 256) sMb[i] = Mb[i];
  if (threadIdx.x < D) { sW1[threadIdx.x] = cW1[threadIdx.x]; sB1[threadIdx.x] = cb1[threadIdx.x]; }
  if (threadIdx.x < H) sC[threadIdx.x] = cconst[threadIdx.x];
  __syncthreads();

  int wid = threadIdx.x >> 5, lane = threadIdx.x & 31;
  int e = blockIdx.x * 8 + wid;
  if (e >= E) return;
  int sN = src[e], tN = tgt[e];

  float curv = 1.0f - aggd[tN] / fmaxf(dist[e], 1e-6f);

  // curvature bias: lane handles j = lane*4..lane*4+3, accumulates all 8 heads
  float bacc[H];
#pragma unroll
  for (int hh = 0; hh < H; ++hh) bacc[hh] = 0.0f;
#pragma unroll
  for (int jj = 0; jj < 4; ++jj) {
    int j = lane * 4 + jj;
    float r = fmaxf(curv * sW1[j] + sB1[j], 0.0f);
#pragma unroll
    for (int hh = 0; hh < H; ++hh) bacc[hh] += r * sMb[j * H + hh];
  }
#pragma unroll
  for (int hh = 0; hh < H; ++hh)
    for (int m = 16; m >= 1; m >>= 1) bacc[hh] += __shfl_xor(bacc[hh], m, 32);

  // per-head qk: lane covers 4 dims of head (lane>>2); reduce within lane-groups of 4
  int dim = lane * 4;
  float4 qv = *(const float4*)(q + (size_t)sN * D + dim);
  float4 kv = *(const float4*)(k + (size_t)tN * D + dim);
  float p = qv.x * kv.x + qv.y * kv.y + qv.z * kv.z + qv.w * kv.w;
  p += __shfl_xor(p, 1, 32);
  p += __shfl_xor(p, 2, 32);
  if ((lane & 3) == 0) {
    int hh = lane >> 2;
    float sc = p * 0.25f + bacc[hh] + sC[hh];   // 1/sqrt(DH)=0.25
    scores[(size_t)e * H + hh] = sc;
    atomicMax(&smax[(size_t)tN * H + hh], enc_f32(sc));
  }
}

// exp + denom per (edge, head); scores overwritten with exp
__global__ void __launch_bounds__(256)
edge_expdenomH_kernel(float* __restrict__ scores, const unsigned int* __restrict__ smax,
                      const int* __restrict__ tgt, float* __restrict__ denomS, int E)
{
  int i = blockIdx.x * 256 + threadIdx.x;
  if (i >= E * H) return;
  int e = i >> 3, hh = i & 7;
  int t = tgt[e];
  float ex = expf(scores[i] - dec_f32(smax[(size_t)t * H + hh]));
  scores[i] = ex;
  atomicAdd(&denomS[(size_t)t * H + hh], ex);
}

// aggm[src] += prob_h * v[tgt]  (wave per edge, 4 dims per lane)
__global__ void __launch_bounds__(256)
edge_msg_kernel(const float* __restrict__ scores, const float* __restrict__ denomS,
                const float* __restrict__ v, const int* __restrict__ src,
                const int* __restrict__ tgt, float* __restrict__ aggm, int E)
{
  int wid = threadIdx.x >> 5, lane = threadIdx.x & 31;
  int e = blockIdx.x * 8 + wid;
  if (e >= E) return;
  int sN = src[e], tN = tgt[e];
  int hh = lane >> 2;
  float prob = scores[(size_t)e * H + hh] / denomS[(size_t)tN * H + hh];
  float4 vv = *(const float4*)(v + (size_t)tN * D + lane * 4);
  float* p = aggm + (size_t)sN * D + lane * 4;
  atomicAdd(p + 0, prob * vv.x);
  atomicAdd(p + 1, prob * vv.y);
  atomicAdd(p + 2, prob * vv.z);
  atomicAdd(p + 3, prob * vv.w);
}

// mean-pool accumulate (wave per node)
__global__ void __launch_bounds__(256)
pool_kernel(const float* __restrict__ h, const int* __restrict__ batch,
            float* __restrict__ pooled, float* __restrict__ counts, int M)
{
  int wid = threadIdx.x >> 5, lane = threadIdx.x & 31;
  int n = blockIdx.x * 8 + wid;
  if (n >= M) return;
  int g = batch[n];
  float4 v = *(const float4*)(h + (size_t)n * D + lane * 4);
  float* p = pooled + (size_t)g * D + lane * 4;
  atomicAdd(p + 0, v.x);
  atomicAdd(p + 1, v.y);
  atomicAdd(p + 2, v.z);
  atomicAdd(p + 3, v.w);
  if (lane == 0) atomicAdd(&counts[g], 1.0f);
}

// final MLP per graph: relu((g/cnt)@W1+b1)@W2+b2
__global__ void final_mlp_kernel(const float* __restrict__ pooled, const float* __restrict__ counts,
                                 const float* __restrict__ W1, const float* __restrict__ b1,
                                 const float* __restrict__ W2, const float* __restrict__ b2,
                                 float* __restrict__ out)
{
  __shared__ float sh[64];
  int g = blockIdx.x, j = threadIdx.x;
  float cnt = fmaxf(counts[g], 1.0f);
  float acc = b1[j];
  for (int d = 0; d < D; ++d) acc += (pooled[(size_t)g * D + d] / cnt) * W1[d * 64 + j];
  sh[j] = fmaxf(acc, 0.0f) * W2[j];
  __syncthreads();
  if (j == 0) {
    float o = b2[0];
    for (int i = 0; i < 64; ++i) o += sh[i];
    out[g] = o;
  }
}

// ---------------- host ----------------
static inline void gemm(hipStream_t s, const float* A, const float* B, const float* bias,
                        const float* resid, float* C, int M, int K, int N, int flags)
{
  dim3 grid(N / 64, (M + 127) / 128);
  wmma_gemm_kernel<<<grid, 256, 0, s>>>(A, B, bias, resid, C, M, K, N, flags);
}

extern "C" void kernel_launch(void* const* d_in, const int* in_sizes, int n_in,
                              void* d_out, int out_size, void* d_ws, size_t ws_size,
                              hipStream_t stream)
{
  (void)in_sizes; (void)n_in; (void)out_size; (void)ws_size;
  const float* x      = (const float*)d_in[0];
  const int*   eidx   = (const int*)  d_in[1];
  const int*   batch  = (const int*)  d_in[2];
  const float* node_W = (const float*)d_in[3];
  const float* node_b = (const float*)d_in[4];
  const float* cW1    = (const float*)d_in[5];
  const float* cb1    = (const float*)d_in[6];
  const float* cW2    = (const float*)d_in[7];
  const float* cb2    = (const float*)d_in[8];
  const float* qW     = (const float*)d_in[9];
  const float* qb     = (const float*)d_in[10];
  const float* kW     = (const float*)d_in[11];
  const float* kb     = (const float*)d_in[12];
  const float* vW     = (const float*)d_in[13];
  const float* vb     = (const float*)d_in[14];
  const float* oW     = (const float*)d_in[15];
  const float* ob     = (const float*)d_in[16];
  const float* bW     = (const float*)d_in[17];
  const float* bb     = (const float*)d_in[18];
  const float* f1W    = (const float*)d_in[19];
  const float* f1b    = (const float*)d_in[20];
  const float* f2W    = (const float*)d_in[21];
  const float* f2b    = (const float*)d_in[22];
  const float* n1s    = (const float*)d_in[23];
  const float* n1b    = (const float*)d_in[24];
  const float* n2s    = (const float*)d_in[25];
  const float* n2b    = (const float*)d_in[26];
  const float* outW1  = (const float*)d_in[27];
  const float* outb1  = (const float*)d_in[28];
  const float* outW2  = (const float*)d_in[29];
  const float* outb2  = (const float*)d_in[30];
  const int* src = eidx;
  const int* tgt = eidx + N_EDGES;

  // workspace carve (bump allocator, 256B aligned)
  char* base = (char*)d_ws;
  size_t off = 0;
  auto wsa = [&](size_t bytes) -> void* {
    void* r = base + off;
    off += (bytes + 255) & ~(size_t)255;
    return r;
  };
  float* h    = (float*)wsa((size_t)N_NODES * D * 4);
  float* xn   = (float*)wsa((size_t)N_NODES * D * 4);
  float* q    = (float*)wsa((size_t)N_NODES * D * 4);
  float* k    = (float*)wsa((size_t)N_NODES * D * 4);
  float* v    = (float*)wsa((size_t)N_NODES * D * 4);
  float* t4   = (float*)wsa((size_t)N_NODES * 4 * D * 4);
  float* sim  = (float*)wsa((size_t)N_EDGES * 4);
  float* dist = (float*)wsa((size_t)N_EDGES * 4);
  float* scr  = (float*)wsa((size_t)N_EDGES * H * 4);
  float* Mb   = (float*)wsa((size_t)D * H * 4);
  float* cc   = (float*)wsa((size_t)H * 4);
  // zeroed-per-layer accumulators (kept contiguous -> one memset)
  size_t z0 = off;
  unsigned int* smaxA  = (unsigned int*)wsa((size_t)N_NODES * 4);
  float*        denomA = (float*)       wsa((size_t)N_NODES * 4);
  float*        aggd   = (float*)       wsa((size_t)N_NODES * 4);
  unsigned int* smaxH  = (unsigned int*)wsa((size_t)N_NODES * H * 4);
  float*        denomH = (float*)       wsa((size_t)N_NODES * H * 4);
  float*        aggm   = (float*)       wsa((size_t)N_NODES * D * 4);
  size_t zbytes = off - z0;
  // pooled region (zeroed once)
  size_t p0 = off;
  float* pooled = (float*)wsa((size_t)G * D * 4);
  float* counts = (float*)wsa((size_t)G * 4);
  size_t pbytes = off - p0;

  const int EB_W = (N_EDGES + 7) / 8;        // wave-per-edge grids
  const int EB_T = (N_EDGES + 255) / 256;    // thread-per-edge grids
  const int EB_H = (N_EDGES * H + 255) / 256;
  const int NB_W = (N_NODES + 7) / 8;

  // node embedding: h = x @ node_W + node_b
  gemm(stream, x, node_W, node_b, nullptr, h, N_NODES, F_IN, D, 0);

  for (int l = 0; l < L; ++l) {
    hipMemsetAsync(base + z0, 0, zbytes, stream);

    layernorm_kernel<<<NB_W, 256, 0, stream>>>(h, n1s + l * D, n1b + l * D, xn, N_NODES);
    gemm(stream, xn, qW + (size_t)l * D * D, qb + l * D, nullptr, q, N_NODES, D, D, 0);
    gemm(stream, xn, kW + (size_t)l * D * D, kb + l * D, nullptr, k, N_NODES, D, D, 0);
    gemm(stream, xn, vW + (size_t)l * D * D, vb + l * D, nullptr, v, N_NODES, D, D, 0);

    // curvature pipeline
    edge_simdist_kernel<<<EB_W, 256, 0, stream>>>(h, src, tgt, sim, dist, smaxA, N_EDGES);
    edge_expdenom_kernel<<<EB_T, 256, 0, stream>>>(sim, smaxA, tgt, denomA, N_EDGES);
    edge_alphaagg_kernel<<<EB_T, 256, 0, stream>>>(sim, dist, denomA, tgt, aggd, N_EDGES);
    prep_curv_proj_kernel<<<4, 256, 0, stream>>>(cW2, cb2, bW + (size_t)l * D * H, bb + l * H, Mb, cc);

    // attention
    edge_scores_kernel<<<EB_W, 256, 0, stream>>>(q, k, src, tgt, aggd, dist,
                                                 cW1, cb1, Mb, cc, scr, smaxH, N_EDGES);
    edge_expdenomH_kernel<<<EB_H, 256, 0, stream>>>(scr, smaxH, tgt, denomH, N_EDGES);
    edge_msg_kernel<<<EB_W, 256, 0, stream>>>(scr, denomH, v, src, tgt, aggm, N_EDGES);

    // h = h + aggm @ oW + ob
    gemm(stream, aggm, oW + (size_t)l * D * D, ob + l * D, h, h, N_NODES, D, D, 2);

    // FFN with residual
    layernorm_kernel<<<NB_W, 256, 0, stream>>>(h, n2s + l * D, n2b + l * D, xn, N_NODES);
    gemm(stream, xn, f1W + (size_t)l * D * 4 * D, f1b + (size_t)l * 4 * D, nullptr, t4, N_NODES, D, 4 * D, 1);
    gemm(stream, t4, f2W + (size_t)l * 4 * D * D, f2b + l * D, h, h, N_NODES, 4 * D, D, 2);
  }

  hipMemsetAsync(base + p0, 0, pbytes, stream);
  pool_kernel<<<NB_W, 256, 0, stream>>>(h, batch, pooled, counts, N_NODES);
  final_mlp_kernel<<<G, 64, 0, stream>>>(pooled, counts, outW1, outb1, outW2, outb2, (float*)d_out);
}
